// Scalar_Attention_13718125543639
// MI455X (gfx1250) — matrix operand order repaired and checked by reading, not compiled, not run on hardware
//
#include <hip/hip_runtime.h>
#include <hip/hip_bf16.h>
#include <math.h>

// ---------------------------------------------------------------------------
// Scalar attention with quantized positional bias, CDNA5 (gfx1250) wave32.
// bias[i,j] = k[i].(tx[ix]+ty[iy]+tz[iz])  ==>  precompute proj_t = K @ T_t^T
// then bias is 3 scalar gathers.  All GEMMs use v_wmma_f32_16x16x32_bf16.
// All operands are laid out so fragment loads are contiguous b128 loads.
// ---------------------------------------------------------------------------

typedef __attribute__((ext_vector_type(16))) __bf16 v16bf;
typedef __attribute__((ext_vector_type(8)))  float  v8f;

#define NPTS   1024
#define DIMS   128
#define QLEN   400
#define LDQKV  384   // qkv stored as 1024 x 384 (q | k | v)
#define SCALE  0.08838834764831845f   // 1/sqrt(128)

// ---- WMMA fragment helpers (A/B: contiguous K per lane half, same order in both, wave32) ----------
// All loads are done as 32-bit words (one word = one packed bf16 pair = one
// fragment VGPR).  Callers guarantee 4-byte alignment (even element offsets).

union frag_bits { v16bf v; unsigned u[8]; };

// A fragment: 16(M) x 32(K) bf16 tile, row-major, leading dim ld (elements).
// lanes 0-15: M=lane, K=0..15 in v0-7
// lanes 16-31: M=lane-16, K=16..31 in v0-7
static __device__ __forceinline__ v16bf load_a_frag(const __bf16* base, int ld) {
  int lane = threadIdx.x & 31;
  int half = lane >> 4;
  int m    = lane & 15;
  const unsigned* row = (const unsigned*)(base + m * ld);
  frag_bits f;
#pragma unroll
  for (int i = 0; i < 4; ++i) {
    f.u[i]     = row[half * 8 + i];       // K = half*16 + 2i, +1
    f.u[4 + i] = row[half * 8 + 4 + i];   // K = half*16 + 8 + 2i, +1
  }
  return f.v;
}

// B fragment fed from a transposed source: X is 16(N) x 32(K) row-major at
// base with leading dim ld; we feed B[k][n] = X[n][k].
// lanes 0-15 hold K=0..15 (vgpr i -> K=2i,2i+1), lanes 16-31 hold K=16..31.
static __device__ __forceinline__ v16bf load_bt_frag(const __bf16* base, int ld) {
  int lane = threadIdx.x & 31;
  int half = lane >> 4;
  int n    = lane & 15;
  const unsigned* row = (const unsigned*)(base + n * ld);
  frag_bits f;
#pragma unroll
  for (int i = 0; i < 8; ++i) f.u[i] = row[half * 8 + i];
  return f.v;
}

static __device__ __forceinline__ v8f wmma_bf16(v16bf a, v16bf b, v8f c) {
  return __builtin_amdgcn_wmma_f32_16x16x32_bf16(
      /*neg_a=*/false, a, /*neg_b=*/false, b,
      /*c_mod=*/(short)0, c, /*reuse_a=*/false, /*reuse_b=*/false);
}

// C/D layout: vgpr r -> M = r + 8*(lane/16), N = lane%16
static __device__ __forceinline__ int c_row(int r) {
  return r + 8 * ((threadIdx.x >> 4) & 1);
}
static __device__ __forceinline__ int c_col() { return threadIdx.x & 15; }

static __device__ __forceinline__ int wave_id() {
  return blockIdx.x * (blockDim.x >> 5) + (threadIdx.x >> 5);
}

// ---- conversion kernels ----------------------------------------------------

__global__ void k_convert(const float* __restrict__ src,
                          __bf16* __restrict__ dst, int n) {
  int i = blockIdx.x * blockDim.x + threadIdx.x;
  if (i < n) dst[i] = (__bf16)src[i];
}

// W_qkv (128x384) -> WT bf16 (384x128):  WT[n][d] = W[d][n]
__global__ void k_wt(const float* __restrict__ W, __bf16* __restrict__ WT) {
  int i = blockIdx.x * blockDim.x + threadIdx.x;
  if (i >= LDQKV * DIMS) return;
  int n = i / DIMS, d = i % DIMS;
  WT[i] = (__bf16)W[d * LDQKV + n];
}

// slice [1] of each (3,QLEN,DIMS) table, bf16, row-major (already the
// transposed-B layout for the proj GEMM: B[k][n] = T[n][k]).
__global__ void k_tables(const float* __restrict__ tx,
                         const float* __restrict__ ty,
                         const float* __restrict__ tz,
                         __bf16* __restrict__ dst) {
  int i = blockIdx.x * blockDim.x + threadIdx.x;
  if (i >= 3 * QLEN * DIMS) return;
  int t = i / (QLEN * DIMS);
  int r = i % (QLEN * DIMS);
  const float* src = (t == 0 ? tx : (t == 1 ? ty : tz)) + QLEN * DIMS; // [1]
  dst[i] = (__bf16)src[r];
}

// ---- qkv = features @ W + b  (1024x128 @ 128x384), also emits V^T ----------

__global__ void k_qkv(const __bf16* __restrict__ featB,
                      const __bf16* __restrict__ WT,
                      const float* __restrict__ bias,
                      __bf16* __restrict__ qkvB,
                      __bf16* __restrict__ VT) {
  int w = wave_id();
  const int NT = LDQKV / 16;            // 24
  int tm = w / NT, tn = w % NT;
  if (tm >= NPTS / 16) return;
  const __bf16* A = featB + tm * 16 * DIMS;
  const __bf16* B = WT + tn * 16 * DIMS;
  v8f c = {};
#pragma unroll
  for (int k = 0; k < DIMS; k += 32) {
    v16bf a = load_a_frag(A + k, DIMS);
    v16bf b = load_bt_frag(B + k, DIMS);
    c = wmma_bf16(a, b, c);
  }
  int n  = c_col();
  int gc = tn * 16 + n;
  float bv = bias[gc];
  if (tn >= (2 * DIMS) / 16) {          // v tile: wave-uniform branch
#pragma unroll
    for (int r = 0; r < 8; ++r) {
      int gr = tm * 16 + c_row(r);
      __bf16 hv = (__bf16)(c[r] + bv);
      qkvB[gr * LDQKV + gc] = hv;
      VT[(size_t)(gc - 2 * DIMS) * NPTS + gr] = hv;   // transposed copy
    }
  } else {
#pragma unroll
    for (int r = 0; r < 8; ++r) {
      int gr = tm * 16 + c_row(r);
      qkvB[gr * LDQKV + gc] = (__bf16)(c[r] + bv);
    }
  }
}

// ---- proj_t = K @ table_t^T  (3 x 1024x400, K=128) -------------------------

__global__ void k_proj(const __bf16* __restrict__ qkvB,
                       const __bf16* __restrict__ Tb,
                       float* __restrict__ proj) {
  int w = wave_id();
  const int NT = QLEN / 16;             // 25
  int t  = w / ((NPTS / 16) * NT);
  int r0 = w % ((NPTS / 16) * NT);
  int tm = r0 / NT, tn = r0 % NT;
  if (t >= 3) return;
  const __bf16* A = qkvB + tm * 16 * LDQKV + DIMS;          // k block
  const __bf16* B = Tb + t * QLEN * DIMS + tn * 16 * DIMS;  // table rows
  v8f c = {};
#pragma unroll
  for (int k = 0; k < DIMS; k += 32) {
    v16bf a = load_a_frag(A + k, LDQKV);
    v16bf b = load_bt_frag(B + k, DIMS);
    c = wmma_bf16(a, b, c);
  }
  float* out = proj + (size_t)t * NPTS * QLEN;
  int n = c_col();
#pragma unroll
  for (int r = 0; r < 8; ++r)
    out[(tm * 16 + c_row(r)) * QLEN + tn * 16 + n] = c[r];
}

// ---- S = (q @ k^T + bias_gather) * scale -----------------------------------

__global__ void k_scores(const __bf16* __restrict__ qkvB,
                         const float* __restrict__ vd,
                         const float* __restrict__ proj,
                         float* __restrict__ S) {
  int w = wave_id();
  int tm = w / (NPTS / 16), tn = w % (NPTS / 16);
  if (tm >= NPTS / 16) return;

  int n = c_col();
  int j = tn * 16 + n;

  // Get the HBM voxel-distance stream (the roofline-dominant traffic)
  // in flight underneath the WMMA dot-product chain.
#pragma unroll
  for (int r = 0; r < 8; ++r) {
    int i = tm * 16 + c_row(r);
    __builtin_prefetch(vd + ((size_t)i * NPTS + j) * 3, 0, 0);
  }

  const __bf16* A  = qkvB + tm * 16 * LDQKV;            // q rows
  const __bf16* Bt = qkvB + tn * 16 * LDQKV + DIMS;     // k rows (transposed)
  v8f c = {};
#pragma unroll
  for (int k = 0; k < DIMS; k += 32) {
    v16bf a = load_a_frag(A + k, LDQKV);
    v16bf b = load_bt_frag(Bt + k, LDQKV);
    c = wmma_bf16(a, b, c);
  }
  const float* px = proj;
  const float* py = proj + (size_t)NPTS * QLEN;
  const float* pz = proj + 2 * (size_t)NPTS * QLEN;
#pragma unroll
  for (int r = 0; r < 8; ++r) {
    int i = tm * 16 + c_row(r);
    const float* d3 = vd + ((size_t)i * NPTS + j) * 3;
    int ix = (int)floorf((d3[0] + 4.0f) * 50.0f);
    int iy = (int)floorf((d3[1] + 4.0f) * 50.0f);
    int iz = (int)floorf((d3[2] + 4.0f) * 50.0f);
    ix = min(max(ix, 0), QLEN - 1);
    iy = min(max(iy, 0), QLEN - 1);
    iz = min(max(iz, 0), QLEN - 1);
    float bias = px[i * QLEN + ix] + py[i * QLEN + iy] + pz[i * QLEN + iz];
    S[(size_t)i * NPTS + j] = (c[r] + bias) * SCALE;
  }
}

// ---- row softmax, emit bf16 P ----------------------------------------------

__global__ void __launch_bounds__(256) k_softmax(const float* __restrict__ S,
                                                 __bf16* __restrict__ P) {
  __shared__ float red[8];
  int row = blockIdx.x;
  int tid = threadIdx.x;                 // 256 threads = 8 waves
  const float* s = S + (size_t)row * NPTS;

  float vals[4];
  float m = -INFINITY;
#pragma unroll
  for (int it = 0; it < 4; ++it) {
    vals[it] = s[tid + it * 256];
    m = fmaxf(m, vals[it]);
  }
#pragma unroll
  for (int off = 16; off; off >>= 1) m = fmaxf(m, __shfl_down(m, off, 32));
  if ((tid & 31) == 0) red[tid >> 5] = m;
  __syncthreads();
  m = red[0];
#pragma unroll
  for (int i = 1; i < 8; ++i) m = fmaxf(m, red[i]);
  __syncthreads();

  float sum = 0.0f;
#pragma unroll
  for (int it = 0; it < 4; ++it) {
    vals[it] = __expf(vals[it] - m);
    sum += vals[it];
  }
#pragma unroll
  for (int off = 16; off; off >>= 1) sum += __shfl_down(sum, off, 32);
  if ((tid & 31) == 0) red[tid >> 5] = sum;
  __syncthreads();
  float tot = 0.0f;
#pragma unroll
  for (int i = 0; i < 8; ++i) tot += red[i];
  float inv = 1.0f / tot;

  __bf16* p = P + (size_t)row * NPTS;
#pragma unroll
  for (int it = 0; it < 4; ++it)
    p[tid + it * 256] = (__bf16)(vals[it] * inv);
}

// ---- out = P @ V  (1024x1024 @ 1024x128), V supplied transposed ------------

__global__ void k_out(const __bf16* __restrict__ P,
                      const __bf16* __restrict__ VT,
                      float* __restrict__ out) {
  int w = wave_id();
  int tm = w / (DIMS / 16), tn = w % (DIMS / 16);
  if (tm >= NPTS / 16) return;
  const __bf16* A = P + (size_t)tm * 16 * NPTS;
  const __bf16* B = VT + (size_t)tn * 16 * NPTS;
  v8f c = {};
#pragma unroll 4
  for (int k = 0; k < NPTS; k += 32) {
    v16bf a = load_a_frag(A + k, NPTS);
    v16bf b = load_bt_frag(B + k, NPTS);
    c = wmma_bf16(a, b, c);
  }
  int n = c_col();
#pragma unroll
  for (int r = 0; r < 8; ++r)
    out[(tm * 16 + c_row(r)) * DIMS + tn * 16 + n] = c[r];
}

// ---------------------------------------------------------------------------

extern "C" void kernel_launch(void* const* d_in, const int* in_sizes, int n_in,
                              void* d_out, int out_size, void* d_ws, size_t ws_size,
                              hipStream_t stream) {
  const float* features = (const float*)d_in[0];   // 1024x128
  const float* vd       = (const float*)d_in[1];   // 1024x1024x3
  const float* W_qkv    = (const float*)d_in[2];   // 128x384
  const float* b_qkv    = (const float*)d_in[3];   // 384
  const float* table_x  = (const float*)d_in[4];   // 3x400x128
  const float* table_y  = (const float*)d_in[5];
  const float* table_z  = (const float*)d_in[6];
  float* out = (float*)d_out;

  char* ws = (char*)d_ws;
  __bf16* featB = (__bf16*)(ws);                   //  256 KiB
  __bf16* WT    = (__bf16*)(ws + 262144);          //   96 KiB (384x128)
  __bf16* Tb    = (__bf16*)(ws + 360448);          //  300 KiB (3x400x128)
  __bf16* qkvB  = (__bf16*)(ws + 667648);          //  768 KiB (1024x384)
  __bf16* VT    = (__bf16*)(ws + 1454080);         //  256 KiB (128x1024)
  float*  proj  = (float*) (ws + 1716224);         // 4800 KiB (3x1024x400)
  float*  S     = (float*) (ws + 6631424);         // 4096 KiB (1024x1024)
  __bf16* P     = (__bf16*)(ws + 10825728);        // 2048 KiB (1024x1024)

  // 1) bf16 conversions / transposes
  k_convert<<<(NPTS * DIMS + 255) / 256, 256, 0, stream>>>(features, featB, NPTS * DIMS);
  k_wt<<<(LDQKV * DIMS + 255) / 256, 256, 0, stream>>>(W_qkv, WT);
  k_tables<<<(3 * QLEN * DIMS + 255) / 256, 256, 0, stream>>>(table_x, table_y, table_z, Tb);

  // 2) qkv projection: 64 x 24 tiles, 8 waves/block
  k_qkv<<<(64 * 24) / 8, 256, 0, stream>>>(featB, WT, b_qkv, qkvB, VT);

  // 3) proj tables: 3 x 64 x 25 tiles
  k_proj<<<(3 * 64 * 25) / 8, 256, 0, stream>>>(qkvB, Tb, proj);

  // 4) scores + gathered bias: 64 x 64 tiles
  k_scores<<<(64 * 64) / 8, 256, 0, stream>>>(qkvB, vd, proj, S);

  // 5) softmax, one block per row
  k_softmax<<<NPTS, 256, 0, stream>>>(S, P);

  // 6) P @ V^T-layout: 64 x 8 tiles
  k_out<<<(64 * 8) / 8, 256, 0, stream>>>(P, VT, out);
}
